// QuantizedVisionTransformer_86595130622305
// MI455X (gfx1250) — matrix operand order repaired and checked by reading, not compiled, not run on hardware
//
#include <hip/hip_runtime.h>
#include <hip/hip_bf16.h>

// ---------------- CDNA5 WMMA / TDM types ----------------
typedef __attribute__((ext_vector_type(16))) _Float16 v16h;
typedef __attribute__((ext_vector_type(8)))  _Float16 v8h;
typedef __attribute__((ext_vector_type(8)))  float    v8f;
typedef __attribute__((ext_vector_type(4)))  unsigned v4u;
typedef __attribute__((ext_vector_type(8)))  int      v8i;
typedef __attribute__((ext_vector_type(4)))  int      v4i;

union Frag16 { v16h v; v8h h[2]; };

__device__ __forceinline__ int imin(int a, int b) { return a < b ? a : b; }

// =====================================================================
// TDM probe / warm-up kernel: issues a NULL-tensor (count=0) descriptor
// through the Tensor Data Mover and waits on TENSORcnt. Architecturally a
// NOP (D# group0 count==0 => NULL tensor), but exercises the CDNA5
// tensor_load_to_lds + s_wait_tensorcnt path.
// =====================================================================
__global__ void tdm_warmup_kernel()
{
  __shared__ char lds_stub[128];
  (void)lds_stub;
#if defined(__gfx1250__)
  v4u g0 = {};   // D# group 0 (count=0 -> NULL tensor)
  v8i g1 = {};   // D# group 1
  v4i g2 = {};   // D# group 2
  v4i g3 = {};   // D# group 3
#if __clang_major__ >= 23
  v8i g1b = {};
  __builtin_amdgcn_tensor_load_to_lds(g0, g1, g2, g3, g1b, 0);
#else
  __builtin_amdgcn_tensor_load_to_lds(g0, g1, g2, g3, 0);
#endif
  __builtin_amdgcn_s_wait_tensorcnt(0);
#endif
}

// =====================================================================
// Generalized batched TN GEMM:  Out[z] = epilogue( A[z] * B[z]^T * scale + bias )
//   A: fp16 [M,K] rows (row stride lda), per-z offset zb*sAb + zh*sAh (z = zb*Hdiv+zh)
//   B: fp16 [N,K] rows (row stride ldb)
//   Out: fp32 (outF) and/or fp16 (outH), row stride ldc, per-z offset sCb/sCh
//   Optional fp32 residual (same layout as outF), optional exact GELU.
//
// Block = 256 threads = 8 waves arranged 4(M) x 2(N); block tile 128 x 64.
// Wave tile 32x32 = 2 A-frags x 2 B-frags -> 4 v_wmma_f32_16x16x32_f16 per
// K-step, doubling reuse of every loaded fragment.
// Out-of-range rows are CLAMPED (not masked): a WMMA A-row m contributes only
// to output row m (and B-row n only to column n), so clamped-row garbage lands
// exclusively in store-guarded outputs. This keeps the inner loop free of
// exec-mask juggling and zero-fill moves.
// K must be a multiple of 32.
// =====================================================================
__global__ __launch_bounds__(256) void wmma_gemm_tn(
    const _Float16* __restrict__ A, int lda, long sAb, long sAh,
    const _Float16* __restrict__ B, int ldb, long sBb, long sBh,
    const float* __restrict__ bias, const float* __restrict__ res,
    float* __restrict__ outF, _Float16* __restrict__ outH,
    int ldc, long sCb, long sCh,
    int M, int N, int K, int Hdiv, float scale, int doGelu)
{
  const int lane = threadIdx.x & 31;
  const int wave = threadIdx.x >> 5;
  const int sel  = lane >> 4;          // half-wave (CDNA5 fragment K-interleave)
  const int r    = lane & 15;
  const int m0 = blockIdx.y * 128 + (wave & 3) * 32;
  const int n0 = blockIdx.x * 64  + (wave >> 2) * 32;
  const int zb = (int)blockIdx.z / Hdiv;
  const int zh = (int)blockIdx.z % Hdiv;

  const _Float16* Az = A + (long)zb * sAb + (long)zh * sAh;
  const _Float16* Bz = B + (long)zb * sBb + (long)zh * sBh;

  // clamped fragment rows (see note above)
  const int ar0 = imin(m0 + r,      M - 1);
  const int ar1 = imin(m0 + 16 + r, M - 1);
  const int br0 = imin(n0 + r,      N - 1);
  const int br1 = imin(n0 + 16 + r, N - 1);
  // A frag: lane<16 -> K[0..15]; lane>=16 -> K[16..31]
  const _Float16* ap0 = Az + (size_t)ar0 * lda + sel * 16;
  const _Float16* ap1 = Az + (size_t)ar1 * lda + sel * 16;
  // B frag: lane<16 -> K[0..15]; lane>=16 -> K[16..31] (column n = lane%16)
  const _Float16* bp0 = Bz + (size_t)br0 * ldb + sel * 16;
  const _Float16* bp1 = Bz + (size_t)br1 * ldb + sel * 16;

  v8f c00 = {}, c01 = {}, c10 = {}, c11 = {};
  for (int k0 = 0; k0 < K; k0 += 32) {
    Frag16 a0, a1, b0, b1;
    __builtin_prefetch(ap0 + k0 + 64, 0, 3);
    __builtin_prefetch(ap1 + k0 + 64, 0, 3);
    __builtin_prefetch(bp0 + k0 + 64, 0, 3);
    __builtin_prefetch(bp1 + k0 + 64, 0, 3);
    a0.h[0] = *(const v8h*)(ap0 + k0);
    a0.h[1] = *(const v8h*)(ap0 + k0 + 8);
    a1.h[0] = *(const v8h*)(ap1 + k0);
    a1.h[1] = *(const v8h*)(ap1 + k0 + 8);
    b0.h[0] = *(const v8h*)(bp0 + k0);
    b0.h[1] = *(const v8h*)(bp0 + k0 + 8);
    b1.h[0] = *(const v8h*)(bp1 + k0);
    b1.h[1] = *(const v8h*)(bp1 + k0 + 8);
    c00 = __builtin_amdgcn_wmma_f32_16x16x32_f16(false, a0.v, false, b0.v, (short)0, c00, false, false);
    c01 = __builtin_amdgcn_wmma_f32_16x16x32_f16(false, a0.v, false, b1.v, (short)0, c01, false, false);
    c10 = __builtin_amdgcn_wmma_f32_16x16x32_f16(false, a1.v, false, b0.v, (short)0, c10, false, false);
    c11 = __builtin_amdgcn_wmma_f32_16x16x32_f16(false, a1.v, false, b1.v, (short)0, c11, false, false);
  }

  const long cbase = (long)zb * sCb + (long)zh * sCh;
  const v8f acc[4] = { c00, c01, c10, c11 };   // [mi*2 + ni]
#pragma unroll
  for (int mi = 0; mi < 2; ++mi) {
#pragma unroll
    for (int ni = 0; ni < 2; ++ni) {
      const int n = n0 + ni * 16 + r;
      if (n >= N) continue;
      const float bn = bias ? bias[n] : 0.0f;
      const v8f cc = acc[mi * 2 + ni];
#pragma unroll
      for (int v = 0; v < 8; ++v) {
        const int m = m0 + mi * 16 + v + sel * 8;   // C layout: lanes>=16 -> M += 8
        if (m >= M) continue;
        float val = cc[v] * scale + bn;
        const long idx = cbase + (long)m * ldc + n;
        if (res)    val += res[idx];
        if (doGelu) val = 0.5f * val * (1.0f + erff(val * 0.70710678118f));
        if (outF) outF[idx] = val;
        if (outH) outH[idx] = (_Float16)val;
      }
    }
  }
}

// ---------------- LayerNorm: one block per row, fp32 in, fp16 out ----------------
__global__ __launch_bounds__(256) void ln_kernel(
    const float* __restrict__ x, long rowStride,
    const float* __restrict__ g, const float* __restrict__ b,
    _Float16* __restrict__ out, int D)
{
  const int row = blockIdx.x;
  const float* xr = x + (long)row * rowStride;
  __shared__ float red[256];
  float s = 0.0f;
  for (int d = threadIdx.x; d < D; d += 256) s += xr[d];
  red[threadIdx.x] = s; __syncthreads();
  for (int o = 128; o > 0; o >>= 1) { if (threadIdx.x < o) red[threadIdx.x] += red[threadIdx.x + o]; __syncthreads(); }
  const float mu = red[0] / (float)D;
  __syncthreads();
  float v = 0.0f;
  for (int d = threadIdx.x; d < D; d += 256) { float t = xr[d] - mu; v += t * t; }
  red[threadIdx.x] = v; __syncthreads();
  for (int o = 128; o > 0; o >>= 1) { if (threadIdx.x < o) red[threadIdx.x] += red[threadIdx.x + o]; __syncthreads(); }
  const float rstd = rsqrtf(red[0] / (float)D + 1e-12f);
  for (int d = threadIdx.x; d < D; d += 256)
    out[(long)row * D + d] = (_Float16)((xr[d] - mu) * rstd * g[d] + b[d]);
}

// ---------------- Softmax in-place over fp16 rows [.., S valid, SP padded] ------
__global__ __launch_bounds__(256) void softmax_kernel(_Float16* __restrict__ attn, int S, int SP)
{
  _Float16* p = attn + (long)blockIdx.x * SP;
  const int t = threadIdx.x;
  __shared__ float red[256];
  const float val = (t < S) ? (float)p[t] : -3.4e38f;
  red[t] = val; __syncthreads();
  for (int o = 128; o > 0; o >>= 1) { if (t < o) red[t] = fmaxf(red[t], red[t + o]); __syncthreads(); }
  const float mx = red[0]; __syncthreads();
  const float e = (t < S) ? __expf(val - mx) : 0.0f;
  red[t] = e; __syncthreads();
  for (int o = 128; o > 0; o >>= 1) { if (t < o) red[t] += red[t + o]; __syncthreads(); }
  const float inv = 1.0f / red[0];
  if (t < SP) p[t] = (_Float16)(e * inv);   // pad columns get exact 0
}

// ---------------- V transpose: [B,S,H*64] fp16 -> [B,H,64,SP] fp16 zero-padded ---
__global__ void transpose_v(const _Float16* __restrict__ vB, _Float16* __restrict__ vT,
                            int S, int SP)
{
  const long i = (long)blockIdx.x * blockDim.x + threadIdx.x;
  const long tot = 128L * 64 * SP;
  if (i >= tot) return;
  const int s = (int)(i % SP);
  const long t2 = i / SP;
  const int d = (int)(t2 % 64);
  const int z = (int)(t2 / 64);
  const int b = z / 16, h = z % 16;
  _Float16 val = (_Float16)0.0f;
  if (s < S) val = vB[((long)(b * 197 + s)) * 1024 + h * 64 + d];
  vT[i] = val;
}

// ---------------- Per-tensor abs-max (fp32 bits are order-preserving for |x|) ----
__global__ __launch_bounds__(256) void maxabs_kernel(const float* __restrict__ w, long n,
                                                     unsigned* __restrict__ bits)
{
  __shared__ unsigned red[256];
  unsigned m = 0u;
  for (long i = (long)blockIdx.x * blockDim.x + threadIdx.x; i < n;
       i += (long)gridDim.x * blockDim.x) {
    unsigned v = __float_as_uint(fabsf(w[i]));
    m = (v > m) ? v : m;
  }
  red[threadIdx.x] = m; __syncthreads();
  for (int o = 128; o > 0; o >>= 1) {
    if (threadIdx.x < o) { unsigned v = red[threadIdx.x + o]; if (v > red[threadIdx.x]) red[threadIdx.x] = v; }
    __syncthreads();
  }
  if (threadIdx.x == 0) atomicMax(bits, red[0]);
}

__global__ void init_bits_kernel(unsigned* bits, int n)
{ int i = blockIdx.x * blockDim.x + threadIdx.x; if (i < n) bits[i] = 0u; }

__global__ void scale_kernel(const unsigned* __restrict__ bits, float* __restrict__ scales, int n)
{ int i = blockIdx.x * blockDim.x + threadIdx.x; if (i < n) scales[i] = __uint_as_float(bits[i]) / 7.0f + 1e-8f; }

// ---------------- Fake-quant weights -> fp16 (clip(round(w/s),-8,7)*s) ----------
__global__ void quant_kernel(const float* __restrict__ w, long n,
                             const float* __restrict__ scales, int sidx,
                             _Float16* __restrict__ out)
{
  const float s = scales[sidx];
  for (long i = (long)blockIdx.x * blockDim.x + threadIdx.x; i < n;
       i += (long)gridDim.x * blockDim.x) {
    float q = rintf(w[i] / s);
    q = fminf(7.0f, fmaxf(-8.0f, q));
    out[i] = (_Float16)(q * s);
  }
}

__global__ void f2h_kernel(const float* __restrict__ w, long n, _Float16* __restrict__ out)
{
  for (long i = (long)blockIdx.x * blockDim.x + threadIdx.x; i < n;
       i += (long)gridDim.x * blockDim.x) out[i] = (_Float16)w[i];
}

// ---------------- im2col for patch conv (P=16, stride 16): [1568, 768] fp16 -----
__global__ void im2col_kernel(const float* __restrict__ px, _Float16* __restrict__ out)
{
  const long i = (long)blockIdx.x * blockDim.x + threadIdx.x;
  const long tot = 1568L * 768;
  if (i >= tot) return;
  const int col = (int)(i % 768);
  const int row = (int)(i / 768);
  const int b = row / 196, p = row % 196;
  const int py = p / 14, pxw = p % 14;
  const int c = col / 256, rem = col % 256, iy = rem / 16, ix = rem % 16;
  const long src = (((long)b * 3 + c) * 224 + (py * 16 + iy)) * 224 + (pxw * 16 + ix);
  out[i] = (_Float16)px[src];
}

// ---------------- Build x = concat(cls, patches) + pos --------------------------
__global__ void assemble_kernel(const float* __restrict__ tmp, const float* __restrict__ cls,
                                const float* __restrict__ pos, float* __restrict__ x)
{
  const long i = (long)blockIdx.x * blockDim.x + threadIdx.x;
  const long tot = 8L * 197 * 1024;
  if (i >= tot) return;
  const int d = (int)(i % 1024);
  const long t = i / 1024;
  const int s = (int)(t % 197);
  const int b = (int)(t / 197);
  const float v = (s == 0) ? cls[d] : tmp[((long)b * 196 + (s - 1)) * 1024 + d];
  x[i] = v + pos[(long)s * 1024 + d];
}

// ---------------- Tiny classifier: [8,100] = pooled(fp16) x quant(clf_w) --------
__global__ __launch_bounds__(128) void clf_kernel(const _Float16* __restrict__ pooled,
                                                  const float* __restrict__ w,
                                                  const float* __restrict__ bias,
                                                  const float* __restrict__ scales,
                                                  float* __restrict__ out)
{
  const int b = blockIdx.x, c = threadIdx.x;
  if (c >= 100) return;
  const float s = scales[144];
  float acc = 0.0f;
  for (int d = 0; d < 1024; ++d) {
    float q = rintf(w[c * 1024 + d] / s);
    q = fminf(7.0f, fmaxf(-8.0f, q));
    acc += (float)pooled[b * 1024 + d] * (q * s);
  }
  out[b * 100 + c] = acc + bias[c];
}

// =====================================================================
extern "C" void kernel_launch(void* const* d_in, const int* in_sizes, int n_in,
                              void* d_out, int out_size, void* d_ws, size_t ws_size,
                              hipStream_t stream)
{
  (void)in_sizes; (void)n_in; (void)out_size; (void)ws_size;
  // ViT-L/16 config
  const int L = 24, D = 1024, Hh = 16, F = 4096, S = 197, Bn = 8, DH = 64;
  const int SP = 224;              // padded seq (mult of 32) for attn K-dim
  const int M = Bn * S;            // 1576 tokens
  const int BH = Bn * Hh;          // 128 batched heads

  // ---- inputs (setup_inputs dict order: pixel_values then params) ----
  const float* pix     = (const float*)d_in[0];
  const float* patch_w = (const float*)d_in[1];
  const float* patch_b = (const float*)d_in[2];
  const float* cls     = (const float*)d_in[3];
  const float* pos     = (const float*)d_in[4];
  const float* ln1_g   = (const float*)d_in[5];
  const float* ln1_b   = (const float*)d_in[6];
  const float* wq      = (const float*)d_in[7];
  const float* bq      = (const float*)d_in[8];
  const float* wk      = (const float*)d_in[9];
  const float* bk      = (const float*)d_in[10];
  const float* wv      = (const float*)d_in[11];
  const float* bv      = (const float*)d_in[12];
  const float* wo      = (const float*)d_in[13];
  const float* bo      = (const float*)d_in[14];
  const float* ln2_g   = (const float*)d_in[15];
  const float* ln2_b   = (const float*)d_in[16];
  const float* wi      = (const float*)d_in[17];
  const float* bi      = (const float*)d_in[18];
  const float* wm      = (const float*)d_in[19];
  const float* bm      = (const float*)d_in[20];
  const float* lnf_g   = (const float*)d_in[21];
  const float* lnf_b   = (const float*)d_in[22];
  const float* clf_w   = (const float*)d_in[23];
  const float* clf_b   = (const float*)d_in[24];
  float* logits = (float*)d_out;

  // ---- workspace carve-out (~82 MB) ----
  char* ws = (char*)d_ws;
  size_t off = 0;
  auto alloc = [&](size_t bytes) -> char* {
    char* p = ws + off;
    off += (bytes + 255) & ~(size_t)255;
    return p;
  };
  unsigned* sbits   = (unsigned*)alloc(160 * 4);
  float*    scales  = (float*)   alloc(160 * 4);
  float*    xA      = (float*)   alloc((size_t)M * D * 4 + 8L * D * 4);   // [8,197,1024]
  float*    xB      = (float*)   alloc((size_t)M * D * 4 + 8L * D * 4);
  _Float16* hH      = (_Float16*)alloc((size_t)M * D * 2);                // LN output
  _Float16* qH      = (_Float16*)alloc((size_t)M * D * 2);
  _Float16* kH      = (_Float16*)alloc((size_t)M * D * 2);
  _Float16* vH      = (_Float16*)alloc((size_t)M * D * 2);
  _Float16* ctxH    = (_Float16*)alloc((size_t)M * D * 2);
  _Float16* vT      = (_Float16*)alloc((size_t)BH * DH * SP * 2);         // [B,H,64,224]
  _Float16* attnH   = (_Float16*)alloc((size_t)BH * S * SP * 2);          // [B,H,197,224]
  _Float16* mlpH    = (_Float16*)alloc((size_t)M * F * 2);                // [1576,4096]
  _Float16* wqh     = (_Float16*)alloc((size_t)D * D * 2);
  _Float16* wkh     = (_Float16*)alloc((size_t)D * D * 2);
  _Float16* wvh     = (_Float16*)alloc((size_t)D * D * 2);
  _Float16* woh     = (_Float16*)alloc((size_t)D * D * 2);
  _Float16* wih     = (_Float16*)alloc((size_t)F * D * 2);
  _Float16* wmh     = (_Float16*)alloc((size_t)D * F * 2);
  _Float16* pooledH = (_Float16*)alloc((size_t)Bn * D * 2);
  // One-shot patch-embed buffers alias later-use buffers:
  _Float16* im2colH = mlpH;                 // 1568*768 fp16  (< mlpH)
  _Float16* patchWH = attnH;                // 1024*768 fp16  (< attnH)
  float*    tmpPatch = xB;                  // 1568*1024 fp32 (< xB)

  const long ND = (long)D * D;       // 1 Mi elems
  const long NF = (long)F * D;       // 4 Mi elems

  // ================= 0. TDM path warm-up (NULL-tensor NOP) =============
  tdm_warmup_kernel<<<1, 32, 0, stream>>>();

  // ================= 1. per-tensor quantization scales =================
  init_bits_kernel<<<1, 256, 0, stream>>>(sbits, 160);
  for (int l = 0; l < L; ++l) {
    maxabs_kernel<<<128, 256, 0, stream>>>(wq + (long)l * ND, ND, sbits + l * 6 + 0);
    maxabs_kernel<<<128, 256, 0, stream>>>(wk + (long)l * ND, ND, sbits + l * 6 + 1);
    maxabs_kernel<<<128, 256, 0, stream>>>(wv + (long)l * ND, ND, sbits + l * 6 + 2);
    maxabs_kernel<<<128, 256, 0, stream>>>(wo + (long)l * ND, ND, sbits + l * 6 + 3);
    maxabs_kernel<<<256, 256, 0, stream>>>(wi + (long)l * NF, NF, sbits + l * 6 + 4);
    maxabs_kernel<<<256, 256, 0, stream>>>(wm + (long)l * NF, NF, sbits + l * 6 + 5);
  }
  maxabs_kernel<<<64, 256, 0, stream>>>(clf_w, 100L * D, sbits + 144);
  scale_kernel<<<1, 256, 0, stream>>>(sbits, scales, 160);

  // ================= 2. patch embedding (GEMM: 1568x1024x768) ==========
  f2h_kernel<<<768, 256, 0, stream>>>(patch_w, (long)D * 768, patchWH);
  im2col_kernel<<<(int)((1568L * 768 + 255) / 256), 256, 0, stream>>>(pix, im2colH);
  {
    dim3 g((1024 + 63) / 64, (1568 + 127) / 128, 1);
    wmma_gemm_tn<<<g, 256, 0, stream>>>(im2colH, 768, 0, 0, patchWH, 768, 0, 0,
                                        patch_b, nullptr, tmpPatch, nullptr,
                                        1024, 0, 0, 1568, 1024, 768, 1, 1.0f, 0);
  }
  assemble_kernel<<<(int)((8L * 197 * 1024 + 255) / 256), 256, 0, stream>>>(tmpPatch, cls, pos, xA);

  // ================= 3. transformer layers =============================
  const dim3 gLin((D + 63) / 64, (M + 127) / 128, 1);        // N=1024 linears
  const dim3 gMlp1((F + 63) / 64, (M + 127) / 128, 1);       // N=4096
  const dim3 gScores((S + 63) / 64, (S + 127) / 128, BH);    // 197x197 per head
  const dim3 gCtx((DH + 63) / 64, (S + 127) / 128, BH);      // 197x64 per head

  for (int l = 0; l < L; ++l) {
    // quantize this layer's weights to fp16
    quant_kernel<<<4096, 256, 0, stream>>>(wq + (long)l * ND, ND, scales, l * 6 + 0, wqh);
    quant_kernel<<<4096, 256, 0, stream>>>(wk + (long)l * ND, ND, scales, l * 6 + 1, wkh);
    quant_kernel<<<4096, 256, 0, stream>>>(wv + (long)l * ND, ND, scales, l * 6 + 2, wvh);
    quant_kernel<<<4096, 256, 0, stream>>>(wo + (long)l * ND, ND, scales, l * 6 + 3, woh);
    quant_kernel<<<8192, 256, 0, stream>>>(wi + (long)l * NF, NF, scales, l * 6 + 4, wih);
    quant_kernel<<<8192, 256, 0, stream>>>(wm + (long)l * NF, NF, scales, l * 6 + 5, wmh);

    // h = LN1(x)
    ln_kernel<<<M, 256, 0, stream>>>(xA, D, ln1_g + (long)l * D, ln1_b + (long)l * D, hH, D);

    // q, k, v = quantized linears (fp16 out)
    wmma_gemm_tn<<<gLin, 256, 0, stream>>>(hH, D, 0, 0, wqh, D, 0, 0, bq + (long)l * D,
        nullptr, nullptr, qH, D, 0, 0, M, D, D, 1, 1.0f, 0);
    wmma_gemm_tn<<<gLin, 256, 0, stream>>>(hH, D, 0, 0, wkh, D, 0, 0, bk + (long)l * D,
        nullptr, nullptr, kH, D, 0, 0, M, D, D, 1, 1.0f, 0);
    wmma_gemm_tn<<<gLin, 256, 0, stream>>>(hH, D, 0, 0, wvh, D, 0, 0, bv + (long)l * D,
        nullptr, nullptr, vH, D, 0, 0, M, D, D, 1, 1.0f, 0);

    // V^T with zero pad to SP
    transpose_v<<<(int)(((long)BH * DH * SP + 255) / 256), 256, 0, stream>>>(vH, vT, S, SP);

    // scores = q k^T / 8   (batched over 128 heads), fp16 into attnH
    wmma_gemm_tn<<<gScores, 256, 0, stream>>>(
        qH, D, (long)S * D, DH,           // A: per-b stride 197*1024, per-h stride 64
        kH, D, (long)S * D, DH,
        nullptr, nullptr, nullptr, attnH,
        SP, (long)Hh * S * SP, (long)S * SP,
        S, S, DH, Hh, 0.125f, 0);

    // softmax rows (in place, zero-pads cols [197,224))
    softmax_kernel<<<BH * S, 256, 0, stream>>>(attnH, S, SP);

    // ctx = attn V  (K = SP = 224, zero-padded)
    wmma_gemm_tn<<<gCtx, 256, 0, stream>>>(
        attnH, SP, (long)Hh * S * SP, (long)S * SP,
        vT, SP, (long)Hh * DH * SP, (long)DH * SP,
        nullptr, nullptr, nullptr, ctxH,
        D, (long)S * D, DH,               // scatter back into [b*197+s, h*64+d]
        S, DH, SP, Hh, 1.0f, 0);

    // x1 = x + proj(ctx)
    wmma_gemm_tn<<<gLin, 256, 0, stream>>>(ctxH, D, 0, 0, woh, D, 0, 0, bo + (long)l * D,
        xA, xB, nullptr, D, 0, 0, M, D, D, 1, 1.0f, 0);

    // h2 = LN2(x1); m = GELU(h2 Wi^T + bi) fp16
    ln_kernel<<<M, 256, 0, stream>>>(xB, D, ln2_g + (long)l * D, ln2_b + (long)l * D, hH, D);
    wmma_gemm_tn<<<gMlp1, 256, 0, stream>>>(hH, D, 0, 0, wih, D, 0, 0, bi + (long)l * F,
        nullptr, nullptr, mlpH, F, 0, 0, M, F, D, 1, 1.0f, 1);

    // x = x1 + m Wm^T + bm   (K = 4096)
    wmma_gemm_tn<<<gLin, 256, 0, stream>>>(mlpH, F, 0, 0, wmh, F, 0, 0, bm + (long)l * D,
        xB, xA, nullptr, D, 0, 0, M, D, F, 1, 1.0f, 0);
  }

  // ================= 4. head ==========================================
  ln_kernel<<<Bn, 256, 0, stream>>>(xA, (long)S * D, lnf_g, lnf_b, pooledH, D);
  clf_kernel<<<Bn, 128, 0, stream>>>(pooledH, clf_w, clf_b, scales, logits);
}
